// MSE_Loss_8907762172086
// MI455X (gfx1250) — 8-bit matrix operand order repaired and checked by reading, compiled, not run on hardware
//
#include <hip/hip_runtime.h>

typedef int   v8i __attribute__((ext_vector_type(8)));
typedef float v4f __attribute__((ext_vector_type(4)));

#define BS_ 32
#define N_ 8
#define HW_ (512*512)
#define WGS_PER_B 32
#define THREADS 256
#define WAVES_PER_WG (THREADS/32)
#define KCHUNK 64
#define CHUNKS (HW_/KCHUNK)
#define WAVES_PER_B (WGS_PER_B*WAVES_PER_WG)
#define ITERS (CHUNKS/WAVES_PER_B)

__device__ __forceinline__ unsigned nib2b(unsigned x) {
  return ((x & 0xFu) * 0x00204081u) & 0x01010101u;
}
__device__ __forceinline__ unsigned pack4(v4f f) {
  unsigned r = (f.x > 0.5f) ? 0x1u : 0u;
  r |= (f.y > 0.5f) ? 0x100u : 0u;
  r |= (f.z > 0.5f) ? 0x10000u : 0u;
  r |= (f.w > 0.5f) ? 0x1000000u : 0u;
  return r;
}

__global__ __launch_bounds__(THREADS) void iou_partial_kernel(
    const int* __restrict__ tgt, const float* __restrict__ pred, int* __restrict__ ws)
{
  const int lane = threadIdx.x & 31;
  const int wave = threadIdx.x >> 5;
  const int b    = blockIdx.x / WGS_PER_B;
  const int wg   = blockIdx.x % WGS_PER_B;
  const int wi   = wg * WAVES_PER_WG + wave;

  const int      r    = lane & 15;
  const int      half = lane >> 4;
  const unsigned sh   = half ? 16u : 0u;
  const bool  isPred  = (r < 8);

  const unsigned selA  = (r == 0) ? 0xFFFFFFFFu : 0u;
  const unsigned onesA = (r == 1) ? 0x01010101u : 0u;
  const unsigned selB  = isPred   ? 0xFFFFFFFFu : 0u;
  const unsigned onesB = (r == 8) ? 0x01010101u : 0u;

  const int* tb = tgt + (size_t)b * HW_;
  const float* pb = pred + ((size_t)b * N_ + (isPred ? r : 0)) * (size_t)HW_;

  const v8i vzero = {0,0,0,0,0,0,0,0};
  v8i c = vzero;

  #pragma unroll 4
  for (int it = 0; it < ITERS; ++it) {
    const int base = (wi + it * WAVES_PER_B) * KCHUNK;

    const int t0 = __builtin_nontemporal_load(tb + base + lane);
    const int t1 = __builtin_nontemporal_load(tb + base + 32 + lane);
    const unsigned m0 = __builtin_amdgcn_ballot_w32(t0 > 0);
    const unsigned m1 = __builtin_amdgcn_ballot_w32(t1 > 0);
    v8i a;
    a[0] = (nib2b(m0 >> (sh +  0)) & selA) | onesA;
    a[1] = (nib2b(m0 >> (sh +  4)) & selA) | onesA;
    a[2] = (nib2b(m0 >> (sh +  8)) & selA) | onesA;
    a[3] = (nib2b(m0 >> (sh + 12)) & selA) | onesA;
    a[4] = (nib2b(m1 >> (sh +  0)) & selA) | onesA;
    a[5] = (nib2b(m1 >> (sh +  4)) & selA) | onesA;
    a[6] = (nib2b(m1 >> (sh +  8)) & selA) | onesA;
    a[7] = (nib2b(m1 >> (sh + 12)) & selA) | onesA;

    const v4f* p0 = (const v4f*)(pb + base + half * 16);
    const v4f* p1 = (const v4f*)(pb + base + 32 + half * 16);
    v4f f0 = __builtin_nontemporal_load(p0 + 0);
    v4f f1 = __builtin_nontemporal_load(p0 + 1);
    v4f f2 = __builtin_nontemporal_load(p0 + 2);
    v4f f3 = __builtin_nontemporal_load(p0 + 3);
    v4f g0 = __builtin_nontemporal_load(p1 + 0);
    v4f g1 = __builtin_nontemporal_load(p1 + 1);
    v4f g2 = __builtin_nontemporal_load(p1 + 2);
    v4f g3 = __builtin_nontemporal_load(p1 + 3);
    v8i bm;
    bm[0] = (pack4(f0) & selB) | onesB;
    bm[1] = (pack4(f1) & selB) | onesB;
    bm[2] = (pack4(f2) & selB) | onesB;
    bm[3] = (pack4(f3) & selB) | onesB;
    bm[4] = (pack4(g0) & selB) | onesB;
    bm[5] = (pack4(g1) & selB) | onesB;
    bm[6] = (pack4(g2) & selB) | onesB;
    bm[7] = (pack4(g3) & selB) | onesB;

    c = __builtin_amdgcn_wmma_i32_16x16x64_iu8(false, a, false, bm, c, false, false);
  }

  __shared__ int s_inter[N_], s_psum[N_], s_tsum;
  if (threadIdx.x < N_) { s_inter[threadIdx.x] = 0; s_psum[threadIdx.x] = 0; }
  if (threadIdx.x == N_) s_tsum = 0;
  __syncthreads();
  if (lane < 8) {
    atomicAdd(&s_inter[lane], c[0]);
    atomicAdd(&s_psum[lane],  c[1]);
  } else if (lane == 8) {
    atomicAdd(&s_tsum, c[0]);
  }
  __syncthreads();
  if (threadIdx.x < N_) {
    atomicAdd(&ws[b * N_ + threadIdx.x],            s_inter[threadIdx.x]);
    atomicAdd(&ws[BS_ * N_ + b * N_ + threadIdx.x], s_psum[threadIdx.x]);
  }
  if (threadIdx.x == N_) atomicAdd(&ws[2 * BS_ * N_ + b], s_tsum);
}

__global__ void zero_ws_kernel(int* __restrict__ ws, int n) {
  int i = blockIdx.x * blockDim.x + threadIdx.x;
  if (i < n) ws[i] = 0;
}

__global__ void finalize_kernel(const float* __restrict__ pred_iou,
                                const int* __restrict__ ws, float* __restrict__ out) {
  int i = threadIdx.x;
  int b = i >> 3;
  int inter = ws[i];
  int uni   = ws[BS_ * N_ + i] + ws[2 * BS_ * N_ + b] - inter;
  float iou = (uni > 0) ? ((float)inter / (float)uni) : 0.0f;
  float d = pred_iou[i] - iou;
  out[i] = d * d;
}

extern "C" void kernel_launch(void* const* d_in, const int* in_sizes, int n_in,
                              void* d_out, int out_size, void* d_ws, size_t ws_size,
                              hipStream_t stream) {
  const float* pred_iou = (const float*)d_in[0];
  const int*   tgt      = (const int*)d_in[1];
  const float* pred     = (const float*)d_in[2];
  int* ws = (int*)d_ws;

  const int wsn = 2 * BS_ * N_ + BS_;
  zero_ws_kernel<<<(wsn + 255) / 256, 256, 0, stream>>>(ws, wsn);
  iou_partial_kernel<<<BS_ * WGS_PER_B, THREADS, 0, stream>>>(tgt, pred, ws);
  finalize_kernel<<<1, BS_ * N_, 0, stream>>>(pred_iou, ws, (float*)d_out);
}
